// GCFM_41755672052587
// MI455X (gfx1250) — matrix operand order repaired and checked by reading, not compiled, not run on hardware
//
#include <hip/hip_runtime.h>

// ---------------------------------------------------------------------------
// ACmix-style block for MI455X (gfx1250, wave32).
// 1x1 convs (dominant FLOPs) -> v_wmma_f32_16x16x32_f16, f32 accumulate.
// Activations pre-transposed to [HW][C] f16 so every WMMA fragment is two
// contiguous global_load_b128 per lane; weights pre-converted to f16.
// Fragments double-buffered across the (fully unrolled) K loop so loads for
// step k+1 issue before step k's WMMAs. Prefetch at WGP scope (L2-resident set).
// Everything else is bandwidth-bound; fused elementwise kernels.
// ---------------------------------------------------------------------------

typedef __attribute__((ext_vector_type(16))) _Float16 v16h;
typedef __attribute__((ext_vector_type(8)))  _Float16 v8h;
typedef __attribute__((ext_vector_type(8)))  float    v8f;

#define HEADS 4
#define DHEAD 64
#define CCH   256
#define HDIM  48
#define WDIM  48
#define HW    (HDIM * WDIM)    // 2304
#define BATCH 2
#define KATT  7
#define KK    (KATT * KATT)    // 49
#define NBH   (BATCH * HEADS)  // 8

__device__ __forceinline__ int refl(int t) {
  // ReflectionPad2d semantics (no edge repeat), valid for t in [-3, 50]
  t = (t < 0) ? -t : t;
  return (t > HDIM - 1) ? (2 * (HDIM - 1) - t) : t;
}

// ---------------------------------------------------------------------------
// Pre-pass: convert weights f32 -> f16 (layout unchanged [co][ci])
// ---------------------------------------------------------------------------
__global__ void cvt_w_kernel(const float* __restrict__ W, _Float16* __restrict__ Wh) {
  const int idx = blockIdx.x * blockDim.x + threadIdx.x;
  if (idx >= CCH * CCH) return;
  Wh[idx] = (_Float16)W[idx];
}

// ---------------------------------------------------------------------------
// Pre-pass: transpose + convert activations: Xt[bi][p][c] = (f16) X[bi][c][p]
// c fastest across threads -> coalesced f16 stores; strided f32 reads stay in L2.
// ---------------------------------------------------------------------------
__global__ void transpose_cvt_kernel(const float* __restrict__ X,
                                     _Float16* __restrict__ Xt) {
  const int idx = blockIdx.x * blockDim.x + threadIdx.x;
  if (idx >= BATCH * CCH * HW) return;
  const int c  = idx % CCH;
  const int p  = (idx / CCH) % HW;
  const int bi = idx / (CCH * HW);
  Xt[idx] = (_Float16)X[((size_t)bi * CCH + c) * HW + p];
}

// ---------------------------------------------------------------------------
// Kernel A: out[bi][co][p] = sum_ci W[co][ci] * X[bi][ci][p] + bias[co]
// One wave -> 16 (co) x 64 (pixel) tile; K loop in steps of 32 with WMMA.
// A fragment layout used here (16-bit A 16x32, not ISA table order): lane m=L&15, half=L>>4,
//   element e -> K = half*16 + e (same order as B)
//   -> one contiguous 16xf16 run from Wh row (two 16B loads).
// B fragment layout (16-bit B 32x16): lane n=L&15, element e -> K = half*16+e
//   -> one contiguous 16xf16 run from Xt row (two 16B loads).
// C/D layout: VGPR r -> row = half*8 + r, col = L&15
// ---------------------------------------------------------------------------
__device__ __forceinline__ v16h load_afrag(const _Float16* Ap, int kb) {
  const v8h alo = *(const v8h*)(Ap + kb);
  const v8h ahi = *(const v8h*)(Ap + kb + 8);
  return __builtin_shufflevector(alo, ahi,
      0, 1, 2, 3, 4, 5, 6, 7, 8, 9, 10, 11, 12, 13, 14, 15);
}

__device__ __forceinline__ v16h load_bfrag(const _Float16* Bp, int kb) {
  const v8h blo = *(const v8h*)(Bp + kb);
  const v8h bhi = *(const v8h*)(Bp + kb + 8);
  return __builtin_shufflevector(blo, bhi,
      0, 1, 2, 3, 4, 5, 6, 7, 8, 9, 10, 11, 12, 13, 14, 15);
}

__global__ __launch_bounds__(256)
void conv1x1_wmma(const _Float16* __restrict__ Xt, const _Float16* __restrict__ Wh,
                  const float* __restrict__ bias, float* __restrict__ out) {
  const int gwave = (blockIdx.x * blockDim.x + threadIdx.x) >> 5;
  const int lane  = threadIdx.x & 31;
  const int NT    = HW / 64;                       // 36 pixel-groups per batch
  const int n64   = gwave % NT;
  const int mt    = (gwave / NT) % (CCH / 16);     // 16 co tiles
  const int bi    = gwave / (NT * (CCH / 16));     // batch

  float* Ob = out + (size_t)bi * CCH * HW;
  const int co0  = mt * 16;
  const int p0   = n64 * 64;
  const int half = lane >> 4;
  const int mn   = lane & 15;

  const _Float16* Ap = Wh + (size_t)(co0 + mn) * CCH + half * 16;
  const _Float16* Bp[4];
#pragma unroll
  for (int t = 0; t < 4; ++t)
    Bp[t] = Xt + ((size_t)bi * HW + p0 + t * 16 + mn) * CCH + half * 16;

  // WGP-scope prefetch (locality 3): pull into all cache levels; set is L2-resident.
  __builtin_prefetch(Ap, 0, 3);
#pragma unroll
  for (int t = 0; t < 4; ++t) __builtin_prefetch(Bp[t], 0, 3);

  v8f acc[4];
#pragma unroll
  for (int t = 0; t < 4; ++t)
#pragma unroll
    for (int r = 0; r < 8; ++r) acc[t][r] = 0.0f;

  // double-buffered fragments: loads for step k+1 issue before step k's WMMAs
  v16h a_cur = load_afrag(Ap, 0);
  v16h b_cur[4];
#pragma unroll
  for (int t = 0; t < 4; ++t) b_cur[t] = load_bfrag(Bp[t], 0);

#pragma unroll
  for (int kb = 0; kb < CCH; kb += 32) {
    v16h a_nxt;
    v16h b_nxt[4];
    if (kb + 32 < CCH) {
      a_nxt = load_afrag(Ap, kb + 32);
#pragma unroll
      for (int t = 0; t < 4; ++t) b_nxt[t] = load_bfrag(Bp[t], kb + 32);
      if (kb + 64 < CCH) {
        __builtin_prefetch(Ap + kb + 64, 0, 3);
        __builtin_prefetch(Bp[0] + kb + 64, 0, 3);
        __builtin_prefetch(Bp[2] + kb + 64, 0, 3);
      }
    }
#pragma unroll
    for (int t = 0; t < 4; ++t)
      acc[t] = __builtin_amdgcn_wmma_f32_16x16x32_f16(
          false, a_cur, false, b_cur[t], (short)0, acc[t], false, false);
    if (kb + 32 < CCH) {
      a_cur = a_nxt;
#pragma unroll
      for (int t = 0; t < 4; ++t) b_cur[t] = b_nxt[t];
    }
  }

#pragma unroll
  for (int t = 0; t < 4; ++t)
#pragma unroll
    for (int r = 0; r < 8; ++r) {
      const int row = co0 + half * 8 + r;
      Ob[(size_t)row * HW + p0 + t * 16 + mn] = acc[t][r] + bias[row];
    }
}

// ---------------------------------------------------------------------------
// Kernel B: positional encoding pe[d, yy, xx] = Wp[d,0]*loc_w + Wp[d,1]*loc_h + bp[d]
// ---------------------------------------------------------------------------
__global__ void pe_kernel(const float* __restrict__ Wp, const float* __restrict__ bp,
                          float* __restrict__ pe) {
  const int idx = blockIdx.x * blockDim.x + threadIdx.x;
  if (idx >= DHEAD * HW) return;
  const int p = idx % HW, d = idx / HW;
  const int yy = p / WDIM, xx = p % WDIM;
  const float lw = -1.0f + 2.0f * xx / (float)(WDIM - 1);
  const float lh = -1.0f + 2.0f * yy / (float)(HDIM - 1);
  pe[idx] = Wp[d * 2 + 0] * lw + Wp[d * 2 + 1] * lh + bp[d];
}

// ---------------------------------------------------------------------------
// Kernel C1: attention logits + softmax over the 7x7 reflect-padded window.
// att[k] = s * ( sum_d q*(K[win] - pe[win]) + sum_d q*pe[center] )
// ---------------------------------------------------------------------------
__global__ void att_softmax_kernel(const float* __restrict__ q_ws,
                                   const float* __restrict__ k_ws,
                                   const float* __restrict__ pe,
                                   float* __restrict__ att) {
  const int idx = blockIdx.x * blockDim.x + threadIdx.x;
  if (idx >= NBH * HW) return;
  const int p = idx % HW, n = idx / HW;
  const int bi = n >> 2, head = n & 3;
  const int yy = p / WDIM, xx = p % WDIM;
  const float scaling = 0.125f;  // 64^-0.5

  const float* Q  = q_ws + ((size_t)bi * CCH + head * DHEAD) * HW + p;
  const float* Kb = k_ws + ((size_t)bi * CCH + head * DHEAD) * HW;

  float c0 = 0.0f;
  for (int d = 0; d < DHEAD; ++d)
    c0 += Q[(size_t)d * HW] * pe[(size_t)d * HW + p];

  float logit[KK];
  float mx = -1e30f;
  for (int i = 0; i < KATT; ++i) {
    const int ry = refl(yy + i - 3);
    for (int j = 0; j < KATT; ++j) {
      const int rx = refl(xx + j - 3);
      const int wp = ry * WDIM + rx;
      float s = 0.0f;
      for (int d = 0; d < DHEAD; ++d)
        s += Q[(size_t)d * HW] * (Kb[(size_t)d * HW + wp] - pe[(size_t)d * HW + wp]);
      const float l = scaling * (s + c0);
      logit[i * KATT + j] = l;
      mx = fmaxf(mx, l);
    }
  }
  float sum = 0.0f;
  for (int k = 0; k < KK; ++k) {
    const float e = __expf(logit[k] - mx);
    logit[k] = e;
    sum += e;
  }
  const float inv = 1.0f / sum;
  for (int k = 0; k < KK; ++k)
    att[((size_t)n * KK + k) * HW + p] = logit[k] * inv;
}

// ---------------------------------------------------------------------------
// Kernel C2: out = 0.5 * sum_k att[n,k,p] * V[n,d,win(k,p)]   (writes d_out)
// ---------------------------------------------------------------------------
__global__ void att_apply_kernel(const float* __restrict__ att,
                                 const float* __restrict__ v_ws,
                                 float* __restrict__ out) {
  const int idx = blockIdx.x * blockDim.x + threadIdx.x;
  if (idx >= BATCH * CCH * HW) return;
  const int p  = idx % HW;
  const int ch = (idx / HW) % CCH;
  const int bi = idx / (CCH * HW);
  const int head = ch / DHEAD;
  const int n = bi * HEADS + head;
  const int yy = p / WDIM, xx = p % WDIM;

  const float* V = v_ws + ((size_t)bi * CCH + ch) * HW;
  const float* A = att + (size_t)n * KK * HW + p;
  float acc = 0.0f;
  for (int i = 0; i < KATT; ++i) {
    const int ry = refl(yy + i - 3);
    for (int j = 0; j < KATT; ++j) {
      const int rx = refl(xx + j - 3);
      acc += A[(size_t)(i * KATT + j) * HW] * V[ry * WDIM + rx];
    }
  }
  out[idx] = 0.5f * acc;
}

// ---------------------------------------------------------------------------
// Kernel D1: fold fc + depthwise conv weights: E[co,c,u,v] = sum_j Wdep[co,j,u,v]*Wfc[j,c]
// ---------------------------------------------------------------------------
__global__ void eff_weight_kernel(const float* __restrict__ Wfc,
                                  const float* __restrict__ Wdep,
                                  float* __restrict__ E) {
  const int idx = blockIdx.x * blockDim.x + threadIdx.x;
  if (idx >= CCH * 12 * 9) return;
  const int uv = idx % 9;
  const int c  = (idx / 9) % 12;
  const int co = idx / 108;
  float s = 0.0f;
  for (int j = 0; j < 9; ++j)
    s += Wdep[(size_t)co * 81 + j * 9 + uv] * Wfc[j * 12 + c];
  E[idx] = s;
}

// ---------------------------------------------------------------------------
// Kernel D2: conv branch via folded weights, accumulates 0.5*conv into d_out.
// qkv[b,c,g,p] = {q|k|v}[b, (c&3)*64 + g, p],  g = co>>2  (grouped conv, 64 groups)
// ---------------------------------------------------------------------------
__global__ void conv_branch_kernel(const float* __restrict__ q_ws,
                                   const float* __restrict__ k_ws,
                                   const float* __restrict__ v_ws,
                                   const float* __restrict__ E,
                                   float* __restrict__ out) {
  const int idx = blockIdx.x * blockDim.x + threadIdx.x;
  if (idx >= BATCH * CCH * HW) return;
  const int p  = idx % HW;
  const int co = (idx / HW) % CCH;
  const int bi = idx / (CCH * HW);
  const int g  = co >> 2;
  const int yy = p / WDIM, xx = p % WDIM;

  const float* Ew = E + (size_t)co * 108;
  float acc = 0.0f;
  for (int c = 0; c < 12; ++c) {
    const float* S = (c < 4) ? q_ws : ((c < 8) ? k_ws : v_ws);
    const int ch = (c & 3) * DHEAD + g;
    const float* Sp = S + ((size_t)bi * CCH + ch) * HW;
    for (int u = 0; u < 3; ++u) {
      const int y2 = yy + u - 1;
      if (y2 < 0 || y2 >= HDIM) continue;
      for (int v = 0; v < 3; ++v) {
        const int x2 = xx + v - 1;
        if (x2 < 0 || x2 >= WDIM) continue;
        acc += Ew[c * 9 + u * 3 + v] * Sp[y2 * WDIM + x2];
      }
    }
  }
  out[idx] += 0.5f * acc;  // d_out already holds 0.5*att (sequential kernels)
}

// ---------------------------------------------------------------------------
extern "C" void kernel_launch(void* const* d_in, const int* in_sizes, int n_in,
                              void* d_out, int out_size, void* d_ws, size_t ws_size,
                              hipStream_t stream) {
  const float* x    = (const float*)d_in[0];
  const float* y    = (const float*)d_in[1];
  const float* z    = (const float*)d_in[2];
  const float* W1   = (const float*)d_in[3];
  const float* b1   = (const float*)d_in[4];
  const float* W2   = (const float*)d_in[5];
  const float* b2   = (const float*)d_in[6];
  const float* W3   = (const float*)d_in[7];
  const float* b3   = (const float*)d_in[8];
  const float* Wp   = (const float*)d_in[9];
  const float* bp   = (const float*)d_in[10];
  const float* Wfc  = (const float*)d_in[11];
  const float* Wdep = (const float*)d_in[12];
  float* out = (float*)d_out;

  // workspace carve-up: f32 intermediates then f16 weight/activation copies (~26 MB)
  float* ws = (float*)d_ws;
  const size_t NQ = (size_t)BATCH * CCH * HW;  // 1,179,648
  float* q_ws   = ws;
  float* k_ws   = q_ws + NQ;
  float* v_ws   = k_ws + NQ;
  float* pe_ws  = v_ws + NQ;                      // DHEAD*HW
  float* att_ws = pe_ws + (size_t)DHEAD * HW;     // NBH*KK*HW
  float* E_ws   = att_ws + (size_t)NBH * KK * HW; // CCH*12*9

  _Float16* h = (_Float16*)(E_ws + (size_t)CCH * 12 * 9);
  _Float16* W1h = h;
  _Float16* W2h = W1h + (size_t)CCH * CCH;
  _Float16* W3h = W2h + (size_t)CCH * CCH;
  _Float16* xT  = W3h + (size_t)CCH * CCH;  // [b][HW][C] f16
  _Float16* yT  = xT + NQ;
  _Float16* zT  = yT + NQ;

  const dim3 blk(256);

  cvt_w_kernel<<<(CCH * CCH + 255) / 256, blk, 0, stream>>>(W1, W1h);
  cvt_w_kernel<<<(CCH * CCH + 255) / 256, blk, 0, stream>>>(W2, W2h);
  cvt_w_kernel<<<(CCH * CCH + 255) / 256, blk, 0, stream>>>(W3, W3h);
  transpose_cvt_kernel<<<(int)((NQ + 255) / 256), blk, 0, stream>>>(x, xT);
  transpose_cvt_kernel<<<(int)((NQ + 255) / 256), blk, 0, stream>>>(y, yT);
  transpose_cvt_kernel<<<(int)((NQ + 255) / 256), blk, 0, stream>>>(z, zT);

  const int gemm_waves = BATCH * (CCH / 16) * (HW / 64);  // 1152 waves
  const dim3 gemm_grid(gemm_waves / 8);                   // 8 waves/block -> 144 blocks
  conv1x1_wmma<<<gemm_grid, blk, 0, stream>>>(xT, W1h, b1, q_ws);
  conv1x1_wmma<<<gemm_grid, blk, 0, stream>>>(yT, W2h, b2, k_ws);
  conv1x1_wmma<<<gemm_grid, blk, 0, stream>>>(zT, W3h, b3, v_ws);

  pe_kernel<<<(DHEAD * HW + 255) / 256, blk, 0, stream>>>(Wp, bp, pe_ws);
  att_softmax_kernel<<<(NBH * HW + 255) / 256, blk, 0, stream>>>(q_ws, k_ws, pe_ws, att_ws);
  att_apply_kernel<<<(int)((NQ + 255) / 256), blk, 0, stream>>>(att_ws, v_ws, out);
  eff_weight_kernel<<<(CCH * 12 * 9 + 255) / 256, blk, 0, stream>>>(Wfc, Wdep, E_ws);
  conv_branch_kernel<<<(int)((NQ + 255) / 256), blk, 0, stream>>>(q_ws, k_ws, v_ws, E_ws, out);
}
